// TBLFAM_6073083756632
// MI455X (gfx1250) — matrix operand order repaired and checked by reading, not compiled, not run on hardware
//
#include <hip/hip_runtime.h>
#include <hip/hip_bf16.h>

// ---------------------------------------------------------------------------
// Problem constants (from reference): B=4, N=1024, C=256, H=8, DH=32, OUT=1024
// ---------------------------------------------------------------------------
#define Bq   4
#define Nt   1024
#define Cc   256
#define Hh   8
#define Dh   32
#define BN   4096            // B*N
#define C3   768             // 3*C

typedef __attribute__((ext_vector_type(16))) __bf16 v16bf;
typedef __attribute__((ext_vector_type(8)))  float  v8f;

// ---------------------------------------------------------------------------
// 16x16 C-tile store helper (CDNA5 C/D layout: VGPR i -> row rbase+i, col=lane&15)
// ---------------------------------------------------------------------------
__device__ __forceinline__ void store_tile(
    float* __restrict__ Cb, v8f acc, int tm, int tn, int ldc,
    const float* __restrict__ bias, int initC, float alpha, int lane)
{
    const int col   = lane & 15;
    const int rbase = (lane >> 4) * 8;
    float* cp = Cb + (long long)(tm + rbase) * ldc + tn + col;
#pragma unroll
    for (int i = 0; i < 8; ++i) {
        float v = alpha * acc[i];
        if (initC) { if (bias) v += bias[tn + col]; }
        else       { v += cp[(long long)i * ldc]; }
        cp[(long long)i * ldc] = v;
    }
}

// ---------------------------------------------------------------------------
// Generic wave32 WMMA GEMM:  C[z] (+)= alpha * A[z] @ op(B[z])  (+ bias)
//   - f32 tensors in memory, converted to bf16 at load, f32 accumulate
//   - 32x32 output tile per wave = 2x2 x (v_wmma_f32_16x16x32_bf16),
//     K stepped by 32; A/B fragments each reused by two WMMAs
//   - TRANSB=0: B stored [K,N] row-major; TRANSB=1: B stored [N,K] (A @ B^T)
//   - C base for batch z:  (z/cDiv)*cOuter + (z%cDiv)*cInner
//     (cDiv=1 -> plain batched; cDiv=H -> per-(b,h) tiles into [B,N,H*DH])
//   - initC=1: C = alpha*acc + bias ; initC=0: C += alpha*acc
// VGPR layouts used here, not the ISA table order (A: lane-half K {0..15}/{16..31};
// B: lane-half K {0..15}/{16..31}, one column per lane).
// ---------------------------------------------------------------------------
template<int TRANSB>
__global__ void __launch_bounds__(32) gemm_wmma_bf16(
    const float* __restrict__ A, const float* __restrict__ Bm,
    float* __restrict__ Cm, const float* __restrict__ bias,
    int K, int lda, int ldb, int ldc,
    long long strideA, long long strideB,
    long long cOuter, int cInner, int cDiv,
    int initC, float alpha)
{
    const int lane = threadIdx.x & 31;
    const int tn = blockIdx.x * 32;
    const int tm = blockIdx.y * 32;
    const int z  = blockIdx.z;

    const float* Ab = A  + (long long)z * strideA;
    const float* Bb = Bm + (long long)z * strideB;
    float*       Cb = Cm + (long long)(z / cDiv) * cOuter
                         + (long long)(z % cDiv) * cInner;

    const int r     = lane & 15;
    const int ahalf = (lane >> 4) * 16;   // A K sub-offset: 0 or 16
    const int bhalf = (lane >> 4) * 16;   // B K sub-offset: 0 or 16

    v8f acc00 = {0.f,0.f,0.f,0.f,0.f,0.f,0.f,0.f};
    v8f acc01 = acc00, acc10 = acc00, acc11 = acc00;

    // Incrementally-advanced operand pointers (no per-iteration 64-bit muls)
    const float* ap0 = Ab + (long long)(tm + r) * lda + ahalf;        // rows tm..tm+15
    const float* ap1 = ap0 + (long long)16 * lda;                     // rows tm+16..tm+31
    const float* bp0;
    const float* bp1;
    if (TRANSB) {
        bp0 = Bb + (long long)(tn + r) * ldb + bhalf;                 // cols tn..tn+15
        bp1 = bp0 + (long long)16 * ldb;                              // cols tn+16..tn+31
    } else {
        bp0 = Bb + (long long)bhalf * ldb + tn + r;
        bp1 = bp0 + 16;
    }

    for (int k0 = 0; k0 < K; k0 += 32) {
        v16bf a0, a1, b0, b1;
#pragma unroll
        for (int j = 0; j < 8; ++j) {
            a0[j]     = (__bf16)ap0[j];
            a0[8 + j] = (__bf16)ap0[8 + j];
            a1[j]     = (__bf16)ap1[j];
            a1[8 + j] = (__bf16)ap1[8 + j];
        }
        if (TRANSB) {
#pragma unroll
            for (int e = 0; e < 16; ++e) {
                b0[e] = (__bf16)bp0[e];
                b1[e] = (__bf16)bp1[e];
            }
            bp0 += 32; bp1 += 32;
        } else {
#pragma unroll
            for (int e = 0; e < 16; ++e) {
                b0[e] = (__bf16)bp0[(long long)e * ldb];
                b1[e] = (__bf16)bp1[(long long)e * ldb];
            }
            bp0 += (long long)32 * ldb; bp1 += (long long)32 * ldb;
        }
        ap0 += 32; ap1 += 32;

        acc00 = __builtin_amdgcn_wmma_f32_16x16x32_bf16(false, a0, false, b0, (short)0, acc00, false, false);
        acc01 = __builtin_amdgcn_wmma_f32_16x16x32_bf16(false, a0, false, b1, (short)0, acc01, false, false);
        acc10 = __builtin_amdgcn_wmma_f32_16x16x32_bf16(false, a1, false, b0, (short)0, acc10, false, false);
        acc11 = __builtin_amdgcn_wmma_f32_16x16x32_bf16(false, a1, false, b1, (short)0, acc11, false, false);
    }

    store_tile(Cb, acc00, tm,      tn,      ldc, bias, initC, alpha, lane);
    store_tile(Cb, acc01, tm,      tn + 16, ldc, bias, initC, alpha, lane);
    store_tile(Cb, acc10, tm + 16, tn,      ldc, bias, initC, alpha, lane);
    store_tile(Cb, acc11, tm + 16, tn + 16, ldc, bias, initC, alpha, lane);
}

// ---------------------------------------------------------------------------
// Extract one of {q,k,v} from a fused QKV buffer [B,N,3,H,DH] into compact
// [B,H,N,DH], optionally dividing by the L2 norm (over DH) of a (possibly
// different) source slice — needed for the faithful k_pre_cls bug.
// Block = 256 threads = 8 warps (one head each); grid = B*N.
// ---------------------------------------------------------------------------
__global__ void extract_head_k(const float* __restrict__ val, int whichV,
                               const float* __restrict__ nrm, int whichN,
                               float* __restrict__ dst)
{
    int bn = blockIdx.x;               // b*N + n
    int h  = threadIdx.x >> 5;
    int d  = threadIdx.x & 31;
    long long base = (long long)bn * C3;
    float v = val[base + whichV * Cc + h * Dh + d];
    float o = v;
    if (nrm) {
        float x  = nrm[base + whichN * Cc + h * Dh + d];
        float ss = x * x;
#pragma unroll
        for (int m = 16; m > 0; m >>= 1) ss += __shfl_xor(ss, m, 32);
        o = v * rsqrtf(ss);
    }
    int b = bn >> 10, n = bn & (Nt - 1);
    dst[(((long long)b * Hh + h) * Nt + n) * Dh + d] = o;
}

// ---------------------------------------------------------------------------
// In-place masked+scaled softmax over one attention row.
// logits[m] = S[b,h,n,m] * score[b,m] * (score[b,m] > score[b,n]-0.1 ? 1 : 0)
// (mask zeroes the logit value, matching the reference exactly).
// grid = B*H*N rows, block = 256.
// ---------------------------------------------------------------------------
__global__ void softmax_attn_k(float* __restrict__ S, const float* __restrict__ score)
{
    int row = blockIdx.x;                       // ((b*H + h)*N + n)
    int b   = row / (Hh * Nt);
    int n   = row % Nt;
    float* p = S + (long long)row * Nt;
    const float* sc = score + b * Nt;
    float sn = sc[n];

    __shared__ float vals[Nt];
    __shared__ float red[256];
    int tid = threadIdx.x;

    float lmax = -3.4e38f;
    for (int i = tid; i < Nt; i += 256) {
        float sm = sc[i];
        float m  = (sm > sn - 0.1f) ? 1.f : 0.f;
        float v  = p[i] * sm * m;
        vals[i]  = v;
        lmax     = fmaxf(lmax, v);
    }
    red[tid] = lmax; __syncthreads();
    for (int s = 128; s > 0; s >>= 1) { if (tid < s) red[tid] = fmaxf(red[tid], red[tid + s]); __syncthreads(); }
    float mx = red[0]; __syncthreads();

    float lsum = 0.f;
    for (int i = tid; i < Nt; i += 256) { float e = __expf(vals[i] - mx); vals[i] = e; lsum += e; }
    red[tid] = lsum; __syncthreads();
    for (int s = 128; s > 0; s >>= 1) { if (tid < s) red[tid] += red[tid + s]; __syncthreads(); }
    float inv = 1.f / red[0];

    for (int i = tid; i < Nt; i += 256) p[i] = vals[i] * inv;
}

// ---------------------------------------------------------------------------
// sim_attn[n,m] = (0.5/H) * sum_h (attn_cls[0,h,n,m] + attn_reg[0,h,n,m])
// ---------------------------------------------------------------------------
__global__ void sim_combine_k(const float* __restrict__ ac,
                              const float* __restrict__ ar,
                              float* __restrict__ out)
{
    long long idx = (long long)blockIdx.x * 256 + threadIdx.x;
    const long long NN2 = (long long)Nt * Nt;
    float s = 0.f;
#pragma unroll
    for (int h = 0; h < Hh; ++h) s += ac[h * NN2 + idx] + ar[h * NN2 + idx];
    out[idx] = s * (0.5f / Hh);
}

// ---------------------------------------------------------------------------
// sim_round2 = mask * softmax(sim_attn) / clip(rowsum, 1e-6)
// mask[n,m] = (sim0[n,m]/H > 0.75).  grid = N rows, block = 256.
// ---------------------------------------------------------------------------
__global__ void sim_round2_k(const float* __restrict__ sa,
                             const float* __restrict__ s0,
                             float* __restrict__ out)
{
    int n = blockIdx.x;
    const float* p = sa + (long long)n * Nt;
    const float* q = s0 + (long long)n * Nt;
    float*       o = out + (long long)n * Nt;

    __shared__ float vals[Nt];
    __shared__ float red[256];
    int tid = threadIdx.x;

    float lmax = -3.4e38f;
    for (int i = tid; i < Nt; i += 256) { float v = p[i]; vals[i] = v; lmax = fmaxf(lmax, v); }
    red[tid] = lmax; __syncthreads();
    for (int s = 128; s > 0; s >>= 1) { if (tid < s) red[tid] = fmaxf(red[tid], red[tid + s]); __syncthreads(); }
    float mx = red[0]; __syncthreads();

    float lsum = 0.f;
    for (int i = tid; i < Nt; i += 256) { float e = __expf(vals[i] - mx); vals[i] = e; lsum += e; }
    red[tid] = lsum; __syncthreads();
    for (int s = 128; s > 0; s >>= 1) { if (tid < s) red[tid] += red[tid + s]; __syncthreads(); }
    float inv = 1.f / red[0]; __syncthreads();

    float lsum2 = 0.f;
    for (int i = tid; i < Nt; i += 256) {
        float m = (q[i] * (1.0f / Hh) > 0.75f) ? vals[i] * inv : 0.f;
        vals[i] = m;
        lsum2 += m;
    }
    red[tid] = lsum2; __syncthreads();
    for (int s = 128; s > 0; s >>= 1) { if (tid < s) red[tid] += red[tid + s]; __syncthreads(); }
    float s2 = fmaxf(red[0], 1e-6f);

    for (int i = tid; i < Nt; i += 256) o[i] = vals[i] / s2;
}

// ---------------------------------------------------------------------------
// Host orchestration
// ---------------------------------------------------------------------------
extern "C" void kernel_launch(void* const* d_in, const int* in_sizes, int n_in,
                              void* d_out, int out_size, void* d_ws, size_t ws_size,
                              hipStream_t stream) {
    (void)in_sizes; (void)n_in; (void)out_size; (void)ws_size;

    const float* x_cls      = (const float*)d_in[0];
    const float* x_reg      = (const float*)d_in[1];
    const float* memory_cls = (const float*)d_in[2];
    const float* memory_reg = (const float*)d_in[3];
    const float* cls_score  = (const float*)d_in[4];
    const float* fg_score   = (const float*)d_in[5];
    const float* W_qkv_cls  = (const float*)d_in[6];
    const float* W_qkv_reg  = (const float*)d_in[7];
    const float* W_qkv_mcls = (const float*)d_in[8];
    const float* W_qkv_mreg = (const float*)d_in[9];
    const float* W1   = (const float*)d_in[10];
    const float* b1   = (const float*)d_in[11];
    const float* W2   = (const float*)d_in[12];
    const float* b2   = (const float*)d_in[13];
    const float* Wuc1 = (const float*)d_in[14];
    const float* buc1 = (const float*)d_in[15];
    const float* Wuc2 = (const float*)d_in[16];
    const float* buc2 = (const float*)d_in[17];
    const float* Wur1 = (const float*)d_in[18];
    const float* bur1 = (const float*)d_in[19];
    const float* Wur2 = (const float*)d_in[20];
    const float* bur2 = (const float*)d_in[21];

    float* ws = (float*)d_ws;
    float* out_feat = (float*)d_out;                      // [B,N,OUT] = 4M floats
    float* out_mcls = out_feat + (long long)BN * 1024;    // [B,N,C]
    float* out_mreg = out_mcls + (long long)BN * Cc;      // [B,N,C]

    const long long NN2 = (long long)Nt * Nt;   // 1,048,576
    const long long NDH = (long long)Nt * Dh;   // 32,768
    const long long BHND = (long long)Bq * Hh * NDH;      // 1,048,576

    long long off = 0;
    auto alloc = [&](long long n) { float* p = ws + off; off += n; return p; };

    float* qkv_cls  = alloc((long long)BN * C3);
    float* qkv_reg  = alloc((long long)BN * C3);
    float* qkv_mcls = alloc((long long)BN * C3);
    float* qkv_mreg = alloc((long long)BN * C3);
    float* q_cls_n      = alloc(BHND);
    float* q_reg_n      = alloc(BHND);
    float* k_pre_cls    = alloc(BHND);
    float* k_pre_reg    = alloc(BHND);
    float* v_cls_raw    = alloc(BHND);
    float* v_cls_n      = alloc(BHND);
    float* v_pre_cls_raw= alloc(BHND);
    float* v_pre_cls_n  = alloc(BHND);
    float* v_reg_raw    = alloc(BHND);
    float* attn_cls = alloc((long long)Bq * Hh * NN2);    // 134 MB
    float* attn_reg = alloc((long long)Bq * Hh * NN2);    // 134 MB
    float* sim0     = alloc(NN2);
    float* sim_attn = alloc(NN2);
    float* simr2    = alloc(NN2);
    float* x_buf    = alloc((long long)BN * Cc);
    float* msa      = alloc((long long)BN * 512);
    float* ssf      = alloc((long long)BN * 512);
    float* mem_x_cls= alloc((long long)BN * Cc);
    float* mem_x_reg= alloc((long long)BN * Cc);
    float* hid_uc   = alloc((long long)BN * 1024);
    float* hid_ur   = alloc((long long)BN * 1024);

    auto gemm = [&](const float* A, const float* Bm, float* C, const float* bias,
                    int M, int Nn, int K, int lda, int ldb, int ldc,
                    long long sA, long long sB,
                    long long cOuter, int cInner, int cDiv,
                    int nbatch, int transB, int initC, float alpha) {
        dim3 g(Nn / 32, M / 32, nbatch), blk(32, 1, 1);
        if (transB)
            gemm_wmma_bf16<1><<<g, blk, 0, stream>>>(A, Bm, C, bias, K, lda, ldb, ldc,
                                                     sA, sB, cOuter, cInner, cDiv, initC, alpha);
        else
            gemm_wmma_bf16<0><<<g, blk, 0, stream>>>(A, Bm, C, bias, K, lda, ldb, ldc,
                                                     sA, sB, cOuter, cInner, cDiv, initC, alpha);
    };

    // ---- 1) QKV projections: [4096,256] @ [256,768] -> [B,N,3,H,DH] ----
    gemm(x_cls,      W_qkv_cls,  qkv_cls,  nullptr, BN, C3, Cc, Cc, C3, C3, 0,0, 0,0,1, 1, 0, 1, 1.f);
    gemm(x_reg,      W_qkv_reg,  qkv_reg,  nullptr, BN, C3, Cc, Cc, C3, C3, 0,0, 0,0,1, 1, 0, 1, 1.f);
    gemm(memory_cls, W_qkv_mcls, qkv_mcls, nullptr, BN, C3, Cc, Cc, C3, C3, 0,0, 0,0,1, 1, 0, 1, 1.f);
    gemm(memory_reg, W_qkv_mreg, qkv_mreg, nullptr, BN, C3, Cc, Cc, C3, C3, 0,0, 0,0,1, 1, 0, 1, 1.f);

    // ---- 2) head extraction + L2 norms -> compact [B,H,N,DH] ----
    extract_head_k<<<BN, 256, 0, stream>>>(qkv_cls,  0, qkv_cls,  0, q_cls_n);
    extract_head_k<<<BN, 256, 0, stream>>>(qkv_reg,  0, qkv_reg,  0, q_reg_n);
    extract_head_k<<<BN, 256, 0, stream>>>(qkv_mreg, 1, qkv_mcls, 1, k_pre_cls);   // faithful bug
    extract_head_k<<<BN, 256, 0, stream>>>(qkv_mreg, 1, qkv_mreg, 1, k_pre_reg);
    extract_head_k<<<BN, 256, 0, stream>>>(qkv_cls,  2, nullptr,  0, v_cls_raw);
    extract_head_k<<<BN, 256, 0, stream>>>(qkv_cls,  2, qkv_cls,  2, v_cls_n);
    extract_head_k<<<BN, 256, 0, stream>>>(qkv_mcls, 2, nullptr,  0, v_pre_cls_raw);
    extract_head_k<<<BN, 256, 0, stream>>>(qkv_mcls, 2, qkv_mcls, 2, v_pre_cls_n);
    extract_head_k<<<BN, 256, 0, stream>>>(qkv_reg,  2, nullptr,  0, v_reg_raw);

    // ---- 3) attention logits: q @ k^T (batched over B*H=32, NT, K=32) ----
    gemm(q_cls_n, k_pre_cls, attn_cls, nullptr, Nt, Nt, Dh, Dh, Dh, Nt, NDH, NDH, NN2, 0, 1, 32, 1, 1, 1.f);
    gemm(q_reg_n, k_pre_reg, attn_reg, nullptr, Nt, Nt, Dh, Dh, Dh, Nt, NDH, NDH, NN2, 0, 1, 32, 1, 1, 1.f);

    // ---- 4) masked-scaled softmax, in place ----
    softmax_attn_k<<<Bq * Hh * Nt, 256, 0, stream>>>(attn_cls, cls_score);
    softmax_attn_k<<<Bq * Hh * Nt, 256, 0, stream>>>(attn_reg, fg_score);

    // ---- 5) x = 0.5*attn_cls@v_pre_cls + 0.5*attn_reg@v_pre_cls -> [B,N,H*DH] ----
    gemm(attn_cls, v_pre_cls_raw, x_buf, nullptr, Nt, Dh, Nt, Nt, Dh, Cc, NN2, NDH, (long long)Nt * Cc, Dh, Hh, 32, 0, 1, 0.5f);
    gemm(attn_reg, v_pre_cls_raw, x_buf, nullptr, Nt, Dh, Nt, Nt, Dh, Cc, NN2, NDH, (long long)Nt * Cc, Dh, Hh, 32, 0, 0, 0.5f);

    // ---- 6) sim0 = sum_h v_cls_n[0,h] @ v_pre_cls_n[0,h]^T  (b=0 only) ----
    for (int h = 0; h < Hh; ++h)
        gemm(v_cls_n + h * NDH, v_pre_cls_n + h * NDH, sim0, nullptr,
             Nt, Nt, Dh, Dh, Dh, Nt, 0, 0, 0, 0, 1, 1, 1, (h == 0) ? 1 : 0, 1.f);

    // ---- 7) sim_attn = mean over h of attn[0] ; 8) round-2 masked softmax ----
    sim_combine_k<<<(int)(NN2 / 256), 256, 0, stream>>>(attn_cls, attn_reg, sim_attn);
    sim_round2_k<<<Nt, 256, 0, stream>>>(sim_attn, sim0, simr2);

    // ---- 9) msa = concat(x, x_ori) @ W1 + b1 (x_ori = v-slice view of qkv_cls) ----
    gemm(x_buf,         W1,            msa, b1,      BN, 512, Cc, Cc, 512, 512, 0,0, 0,0,1, 1, 0, 1, 1.f);
    gemm(qkv_cls + 512, W1 + 256*512,  msa, nullptr, BN, 512, Cc, C3, 512, 512, 0,0, 0,0,1, 1, 0, 0, 1.f);

    // ---- 10) soft_sim_feature = sim_round2 @ msa[b]  (A shared across batch) ----
    gemm(simr2, msa, ssf, nullptr, Nt, 512, Nt, Nt, 512, 512, 0, (long long)Nt * 512, (long long)Nt * 512, 0, 1, 4, 0, 1, 1.f);

    // ---- 11) out_feat = concat(ssf, msa) @ W2 + b2  -> d_out ----
    gemm(ssf, W2,             out_feat, b2,      BN, 1024, 512, 512, 1024, 1024, 0,0, 0,0,1, 1, 0, 1, 1.f);
    gemm(msa, W2 + 512*1024,  out_feat, nullptr, BN, 1024, 512, 512, 1024, 1024, 0,0, 0,0,1, 1, 0, 0, 1.f);

    // ---- 12) memory-path attention outputs ----
    gemm(attn_cls, v_cls_raw, mem_x_cls, nullptr, Nt, Dh, Nt, Nt, Dh, Cc, NN2, NDH, (long long)Nt * Cc, Dh, Hh, 32, 0, 1, 1.f);
    gemm(attn_reg, v_reg_raw, mem_x_reg, nullptr, Nt, Dh, Nt, Nt, Dh, Cc, NN2, NDH, (long long)Nt * Cc, Dh, Hh, 32, 0, 1, 1.f);

    // ---- 13) memory_cls_ = (concat(mem_x_cls, v_pre_cls-view) @ Wuc1 + buc1) @ Wuc2 + buc2 ----
    gemm(mem_x_cls,      Wuc1,            hid_uc, buc1,    BN, 1024, Cc,  Cc,  1024, 1024, 0,0, 0,0,1, 1, 0, 1, 1.f);
    gemm(qkv_mcls + 512, Wuc1 + 256*1024, hid_uc, nullptr, BN, 1024, Cc,  C3,  1024, 1024, 0,0, 0,0,1, 1, 0, 0, 1.f);
    gemm(hid_uc,         Wuc2,            out_mcls, buc2,  BN, Cc,  1024, 1024, Cc,  Cc,   0,0, 0,0,1, 1, 0, 1, 1.f);

    // ---- 14) memory_reg_ ----
    gemm(mem_x_reg,      Wur1,            hid_ur, bur1,    BN, 1024, Cc,  Cc,  1024, 1024, 0,0, 0,0,1, 1, 0, 1, 1.f);
    gemm(qkv_mreg + 512, Wur1 + 256*1024, hid_ur, nullptr, BN, 1024, Cc,  C3,  1024, 1024, 0,0, 0,0,1, 1, 0, 0, 1.f);
    gemm(hid_ur,         Wur2,            out_mreg, bur2,  BN, Cc,  1024, 1024, Cc,  Cc,   0,0, 0,0,1, 1, 0, 1, 1.f);
}
